// StateBlock_38792144617953
// MI455X (gfx1250) — matrix operand order repaired and checked by reading, not compiled, not run on hardware
//
#include <hip/hip_runtime.h>
#include <hip/hip_bf16.h>
#include <math.h>

// ---------------------------------------------------------------------------
// MI455X (gfx1250) implementation.
//   * All 4 GEMMs (inj / readout / mlp1 / mlp2) run on v_wmma_f32_16x16x32_bf16
//     with fp32 accumulation.
//   * GEMM tiles are staged memory->LDS with GLOBAL_LOAD_ASYNC_TO_LDS_B128
//     (ASYNCcnt-tracked DMA, no VGPR round trip) into double-buffered LDS,
//     overlapping the next K-tile's loads with the current tile's WMMAs.
//   * The T=1024 sequential scan only advances the S=128 recurrent state;
//     the expensive readout projection is hoisted into a parallel WMMA GEMM.
// ---------------------------------------------------------------------------

typedef __bf16 bf16_t;
typedef bf16_t v16bf __attribute__((ext_vector_type(16)));
typedef float  v8f   __attribute__((ext_vector_type(8)));

#if __has_builtin(__builtin_amdgcn_global_load_async_to_lds_b128) && \
    __has_builtin(__builtin_amdgcn_s_wait_asynccnt)
#define USE_ASYNC_LDS 1
#endif

// The async-LDS builtin takes int4 pointers in AS(1)/AS(3)
// (clang prints AS(1) as `__device__` in HIP mode).
typedef int v4i_t __attribute__((vector_size(16)));
typedef __attribute__((address_space(1))) v4i_t g_v4i;
typedef __attribute__((address_space(3))) v4i_t l_v4i;

#define TILE_M 128
#define TILE_N 128
#define TILE_K 32
#define DIM    1024
#define SDIM   128
#define TSTEPS 1024
#define NROWS  32768   // B*T

union FragBF { v16bf v; unsigned u[8]; };

__device__ inline float gelu_exact(float x) {
    return 0.5f * x * (1.0f + erff(x * 0.70710678118654752440f));
}

// 16-byte global->LDS chunk copy: async DMA on gfx1250 toolchains that expose
// the builtin, synchronous uint4 bounce otherwise.
__device__ inline void copy16_g2l(const bf16_t* __restrict__ g,
                                  unsigned short* l) {
#ifdef USE_ASYNC_LDS
    __builtin_amdgcn_global_load_async_to_lds_b128((g_v4i*)g, (l_v4i*)l, 0, 0);
#else
    *(uint4*)l = *(const uint4*)g;
#endif
}

__device__ inline void copy16_wait() {
#ifdef USE_ASYNC_LDS
    __builtin_amdgcn_s_wait_asynccnt(0);
#endif
}

// ---------------------------------------------------------------------------
// f32 -> bf16 conversion (weights)
// ---------------------------------------------------------------------------
__global__ __launch_bounds__(256) void f32_to_bf16(const float* __restrict__ in,
                                                   bf16_t* __restrict__ out, int n) {
    int i = blockIdx.x * 256 + threadIdx.x;
    if (i < n) out[i] = (bf16_t)in[i];
}

// ---------------------------------------------------------------------------
// Row LayerNorm over DIM=1024.  addInput=0: out = LN(x) (fp32 + bf16 copies)
//                               addInput=1: out = x + LN(x)   (final epilogue)
// ---------------------------------------------------------------------------
__global__ __launch_bounds__(256) void ln_kernel(const float* __restrict__ x,
                                                 const float* __restrict__ w,
                                                 const float* __restrict__ b,
                                                 float* __restrict__ outF,
                                                 bf16_t* __restrict__ outB,
                                                 int addInput) {
    __shared__ float red[256];
    const size_t row = blockIdx.x;
    const float* xr = x + row * DIM;
    const int t = threadIdx.x;

    float v[4];
    float s = 0.0f;
    #pragma unroll
    for (int i = 0; i < 4; ++i) { v[i] = xr[t + 256 * i]; s += v[i]; }
    red[t] = s; __syncthreads();
    for (int off = 128; off > 0; off >>= 1) {
        if (t < off) red[t] += red[t + off];
        __syncthreads();
    }
    const float mu = red[0] * (1.0f / DIM);
    __syncthreads();

    s = 0.0f;
    #pragma unroll
    for (int i = 0; i < 4; ++i) { float d = v[i] - mu; s += d * d; }
    red[t] = s; __syncthreads();
    for (int off = 128; off > 0; off >>= 1) {
        if (t < off) red[t] += red[t + off];
        __syncthreads();
    }
    const float rstd = rsqrtf(red[0] * (1.0f / DIM) + 1e-5f);

    #pragma unroll
    for (int i = 0; i < 4; ++i) {
        const int d = t + 256 * i;
        float ln = (v[i] - mu) * rstd * w[d] + b[d];
        float o  = addInput ? (v[i] + ln) : ln;
        if (outF) outF[row * DIM + d] = o;
        if (outB) outB[row * DIM + d] = (bf16_t)o;
    }
}

// ---------------------------------------------------------------------------
// Generic WMMA GEMM:  C[M,N] = act(A[M,K] @ W[N,K]^T + bias) (+ residual)
// A, W bf16 (K-contiguous).  256 threads = 8 waves; block tile 128x128x32;
// wave tile 32x64 (2x4 16x16 accumulators).  Double-buffered async LDS tiles.
// ---------------------------------------------------------------------------
__global__ __launch_bounds__(256) void gemm_bf16(
    const bf16_t* __restrict__ A, const bf16_t* __restrict__ W,
    const float* __restrict__ bias, const float* __restrict__ residual,
    float* __restrict__ outF, bf16_t* __restrict__ outB,
    int M, int N, int K, int act) {

    __shared__ __align__(16) unsigned short As[2][TILE_M * TILE_K];
    __shared__ __align__(16) unsigned short Bs[2][TILE_N * TILE_K];

    const int tid   = threadIdx.x;
    const int lane  = tid & 31;
    const int wave  = tid >> 5;
    const int waveM = wave & 3;    // 4 wave-rows  * 32
    const int waveN = wave >> 2;   // 2 wave-cols  * 64
    const int mBase = blockIdx.y * TILE_M;
    const int nBase = blockIdx.x * TILE_N;
    const int fidx  = lane & 15;   // fragment M (A) / N (B,C) index
    const int hl    = lane >> 4;   // half-wave select

    // each thread stages two 16B chunks per tile (512 chunks / 256 threads)
    const int c0row = tid >> 2;
    const int c0kk  = (tid & 3) * 8;
    const int c1row = (tid + 256) >> 2;
    const int c1kk  = c0kk;        // (tid+256)&3 == tid&3

    v8f acc[2][4];
    for (int i = 0; i < 2; ++i)
        for (int j = 0; j < 4; ++j)
            for (int r = 0; r < 8; ++r) acc[i][j][r] = 0.0f;

    const int nk = K / TILE_K;

    auto load_tile = [&](int k0, int buf) {
        copy16_g2l(A + (size_t)(mBase + c0row) * K + k0 + c0kk,
                   &As[buf][c0row * TILE_K + c0kk]);
        copy16_g2l(A + (size_t)(mBase + c1row) * K + k0 + c1kk,
                   &As[buf][c1row * TILE_K + c1kk]);
        copy16_g2l(W + (size_t)(nBase + c0row) * K + k0 + c0kk,
                   &Bs[buf][c0row * TILE_K + c0kk]);
        copy16_g2l(W + (size_t)(nBase + c1row) * K + k0 + c1kk,
                   &Bs[buf][c1row * TILE_K + c1kk]);
    };

    load_tile(0, 0);
    copy16_wait();
    __syncthreads();

    for (int kt = 0; kt < nk; ++kt) {
        const int cur = kt & 1;
        if (kt + 1 < nk) load_tile((kt + 1) * TILE_K, 1 - cur);
        if (kt + 2 < nk) {   // warm L2 for the tile after next
            __builtin_prefetch(A + (size_t)(mBase + (tid >> 1)) * K + (kt + 2) * TILE_K, 0, 0);
            __builtin_prefetch(W + (size_t)(nBase + (tid >> 1)) * K + (kt + 2) * TILE_K, 0, 0);
        }

        // A fragment (16-bit A 16x32 layout): lane m = fidx,
        //   r<4 : K = hl*8 + 2r ; r>=4 : K = 16 + hl*8 + 2(r-4)
        FragBF af[2];
        #pragma unroll
        for (int s = 0; s < 2; ++s) {
            int row = waveM * 32 + s * 16 + fidx;
            #pragma unroll
            for (int r = 0; r < 8; ++r) {
                int kk = (r < 4) ? (hl * 8 + 2 * r) : (16 + hl * 8 + 2 * (r - 4));
                af[s].u[r] = *(const unsigned*)&As[cur][row * TILE_K + kk];
            }
        }
        // B fragment (32x16 layout): lane n = fidx, same K order as A fragment
        FragBF bfr[4];
        #pragma unroll
        for (int t = 0; t < 4; ++t) {
            int row = waveN * 64 + t * 16 + fidx;
            #pragma unroll
            for (int r = 0; r < 8; ++r) {
                int kk = (r < 4) ? (hl * 8 + 2 * r) : (16 + hl * 8 + 2 * (r - 4));
                bfr[t].u[r] = *(const unsigned*)&Bs[cur][row * TILE_K + kk];
            }
        }
        #pragma unroll
        for (int s = 0; s < 2; ++s)
            #pragma unroll
            for (int t = 0; t < 4; ++t)
                acc[s][t] = __builtin_amdgcn_wmma_f32_16x16x32_bf16(
                    false, af[s].v, false, bfr[t].v, (short)0, acc[s][t],
                    false, false);

        copy16_wait();       // next tile fully landed in the other buffer
        __syncthreads();     // and everyone is done reading this buffer
    }

    // Epilogue. C layout: lane gives N (= fidx); VGPR r -> M = 8*hl + r.
    for (int t = 0; t < 4; ++t) {
        int gcol = nBase + waveN * 64 + t * 16 + fidx;
        float bv = bias ? bias[gcol] : 0.0f;
        for (int s = 0; s < 2; ++s) {
            #pragma unroll
            for (int r = 0; r < 8; ++r) {
                int grow = mBase + waveM * 32 + s * 16 + hl * 8 + r;
                size_t idx = (size_t)grow * N + gcol;
                float v = acc[s][t][r] + bv;
                if (act == 1) v = gelu_exact(v);
                if (residual) v += residual[idx];
                if (outF) outF[idx] = v;
                if (outB) outB[idx] = (bf16_t)v;
            }
        }
    }
}

// ---------------------------------------------------------------------------
// Sequential recurrence. One workgroup per batch element (32 WGs, 128 thr).
// gate_w / state_flow staged in LDS with padded row stride 132 (bank spread).
// Emits only h_t (bf16) — readout projection is a later WMMA GEMM.
// ---------------------------------------------------------------------------
__global__ __launch_bounds__(128) void scan_kernel(
    const float* __restrict__ gate_w, const float* __restrict__ state_flow,
    const float* __restrict__ gate_b, const float* __restrict__ injF,
    bf16_t* __restrict__ hseq) {

    extern __shared__ float sm[];
    float* gw = sm;                    // 128 * 132
    float* sf = gw + SDIM * 132;       // 128 * 132
    float* h  = sf + SDIM * 132;       // 128

    const int b = blockIdx.x;
    const int s = threadIdx.x;

    for (int i = s; i < SDIM * SDIM; i += SDIM) {
        int r = i >> 7, c = i & 127;
        gw[r * 132 + c] = gate_w[i];
        sf[r * 132 + c] = state_flow[i];
    }
    h[s] = 0.0f;
    const float gb = gate_b[s];
    const float* gr = &gw[s * 132];
    const float* fr = &sf[s * 132];
    __syncthreads();

    for (int t = 0; t < TSTEPS; ++t) {
        const size_t row = (size_t)b * TSTEPS + t;
        float a = 0.0f, f = 0.0f;
        #pragma unroll 8
        for (int k = 0; k < SDIM; ++k) {
            float hk = h[k];                 // LDS broadcast
            a += gr[k] * hk;
            f += fr[k] * hk;
        }
        float g  = 1.0f / (1.0f + expf(-(a + gb + injF[row * SDIM + s])));
        float hn = (1.0f - g) * h[s] + g * f;
        __syncthreads();
        h[s] = hn;
        hseq[row * SDIM + s] = (bf16_t)hn;
        __syncthreads();
    }
}

// ---------------------------------------------------------------------------
extern "C" void kernel_launch(void* const* d_in, const int* in_sizes, int n_in,
                              void* d_out, int out_size, void* d_ws, size_t ws_size,
                              hipStream_t stream) {
    const float* x          = (const float*)d_in[0];
    const float* ln1_w      = (const float*)d_in[1];
    const float* ln1_b      = (const float*)d_in[2];
    const float* state_flow = (const float*)d_in[3];
    const float* injection  = (const float*)d_in[4];
    const float* readout    = (const float*)d_in[5];
    const float* gate_w     = (const float*)d_in[6];
    const float* gate_b     = (const float*)d_in[7];
    const float* mlp_w1     = (const float*)d_in[8];
    const float* mlp_b1     = (const float*)d_in[9];
    const float* mlp_w2     = (const float*)d_in[10];
    const float* mlp_b2     = (const float*)d_in[11];
    const float* ln2_w      = (const float*)d_in[12];
    const float* ln2_b      = (const float*)d_in[13];

    char* ws = (char*)d_ws;
    float*  xnF   = (float*) (ws + 0ull);
    bf16_t* xnB   = (bf16_t*)(ws + 134217728ull);
    float*  xrF   = (float*) (ws + 201326592ull);
    bf16_t* xrB   = (bf16_t*)(ws + 335544320ull);
    float*  yF    = (float*) (ws + 402653184ull);
    bf16_t* hqB   = (bf16_t*)(ws + 536870912ull);
    float*  injF  = (float*) (ws + 805306368ull);
    bf16_t* hsB   = (bf16_t*)(ws + 822083584ull);
    bf16_t* wInjB = (bf16_t*)(ws + 830472192ull);
    bf16_t* wRoB  = (bf16_t*)(ws + 830734336ull);
    bf16_t* w1B   = (bf16_t*)(ws + 830996480ull);
    bf16_t* w2B   = (bf16_t*)(ws + 839385088ull);

    // 1) weights -> bf16
    f32_to_bf16<<<dim3(512),   dim3(256), 0, stream>>>(injection, wInjB, SDIM * DIM);
    f32_to_bf16<<<dim3(512),   dim3(256), 0, stream>>>(readout,   wRoB,  DIM * SDIM);
    f32_to_bf16<<<dim3(16384), dim3(256), 0, stream>>>(mlp_w1,    w1B,   4 * DIM * DIM);
    f32_to_bf16<<<dim3(16384), dim3(256), 0, stream>>>(mlp_w2,    w2B,   4 * DIM * DIM);

    // 2) xn = LN1(x)  (fp32 for residual, bf16 for WMMA)
    ln_kernel<<<dim3(NROWS), dim3(256), 0, stream>>>(x, ln1_w, ln1_b, xnF, xnB, 0);

    // 3) inj[row,s] = xn @ injection^T       (32768 x 128 x K=1024)
    gemm_bf16<<<dim3(1, 256), dim3(256), 0, stream>>>(
        xnB, wInjB, nullptr, nullptr, injF, nullptr, NROWS, SDIM, DIM, 0);

    // 4) recurrence -> h sequence (bf16)
    const size_t scanLds = (size_t)(2 * SDIM * 132 + SDIM) * sizeof(float);
    scan_kernel<<<dim3(32), dim3(128), scanLds, stream>>>(
        gate_w, state_flow, gate_b, injF, hsB);

    // 5) xr = xn + hseq @ readout^T          (32768 x 1024 x K=128)
    gemm_bf16<<<dim3(8, 256), dim3(256), 0, stream>>>(
        hsB, wRoB, nullptr, xnF, xrF, xrB, NROWS, DIM, SDIM, 0);

    // 6) hq = gelu(xr @ w1^T + b1)           (32768 x 4096 x K=1024)
    gemm_bf16<<<dim3(32, 256), dim3(256), 0, stream>>>(
        xrB, w1B, mlp_b1, nullptr, nullptr, hqB, NROWS, 4 * DIM, DIM, 1);

    // 7) y = hq @ w2^T + b2                  (32768 x 1024 x K=4096)
    gemm_bf16<<<dim3(8, 256), dim3(256), 0, stream>>>(
        hqB, w2B, mlp_b2, nullptr, yF, nullptr, NROWS, DIM, 4 * DIM, 0);

    // 8) out = y + LN2(y)
    ln_kernel<<<dim3(NROWS), dim3(256), 0, stream>>>(
        yF, ln2_w, ln2_b, (float*)d_out, nullptr, 1);
}
